// MultiHeadAttention_4690104287376
// MI455X (gfx1250) — matrix operand order repaired and checked by reading, not compiled, not run on hardware
//
#include <hip/hip_runtime.h>

// ---------------------------------------------------------------------------
// MI455X (gfx1250) multi-head attention, bf16 WMMA pipeline.
// Shapes fixed: B=4, N=1024, D_MODEL=1024, H=16, D_V=64.
// ---------------------------------------------------------------------------

#define B_   4
#define N_   1024
#define D_   1024
#define H_   16
#define DV_  64

typedef __bf16 bf16_t;
typedef bf16_t v16bf __attribute__((ext_vector_type(16)));
typedef float  v8f   __attribute__((ext_vector_type(8)));
typedef unsigned int v8u __attribute__((ext_vector_type(8)));

__device__ __forceinline__ unsigned short f2bf(float f) {
    unsigned int u = __builtin_bit_cast(unsigned int, f);
    u += 0x7FFFu + ((u >> 16) & 1u);           // round-to-nearest-even
    return (unsigned short)(u >> 16);
}

__device__ __forceinline__ v8f vzero8() {
    v8f z;
    #pragma unroll
    for (int i = 0; i < 8; ++i) z[i] = 0.0f;
    return z;
}

__device__ __forceinline__ v16bf mk_frag(uint4 lo, uint4 hi) {
    v8u u;
    u[0] = lo.x; u[1] = lo.y; u[2] = lo.z; u[3] = lo.w;
    u[4] = hi.x; u[5] = hi.y; u[6] = hi.z; u[7] = hi.w;
    return __builtin_bit_cast(v16bf, u);
}

__device__ __forceinline__ v8f wmma_bf16(v16bf a, v16bf b, v8f c) {
    // (neg_a, A, neg_b, B, c_mod, C, reuse_a, reuse_b)
    return __builtin_amdgcn_wmma_f32_16x16x32_bf16(false, a, false, b,
                                                   (short)0, c, false, false);
}

// 16-bit A 16x32 fragment from a row-major slab with leading dim ld (elements).
// Caller must fold the K offset into `base`. lane: r = lane&15 (row),
// half = lane>>4. K = half*16 + e contiguous.
__device__ __forceinline__ v16bf load_a_frag(const unsigned short* base, int row,
                                             int half, int ld) {
    const unsigned short* p = base + (size_t)row * ld + half * 16;
    uint4 lo = *(const uint4*)p;
    uint4 hi = *(const uint4*)(p + 8);
    return mk_frag(lo, hi);
}

// 16-bit B 32x16 fragment when B-columns are CONTIGUOUS in memory with leading
// dim ld between columns: lane holds col (lane&15), K = half*16 + e contiguous.
// Caller must fold the K offset into `base`.
__device__ __forceinline__ v16bf load_b_frag_contig(const unsigned short* base,
                                                    int col, int half, int ld) {
    const unsigned short* p = base + (size_t)col * ld + half * 16;
    uint4 lo = *(const uint4*)p;
    uint4 hi = *(const uint4*)(p + 8);
    return mk_frag(lo, hi);
}

// ---------------------------------------------------------------------------
// f32 -> bf16 conversion (vectorized x4)
// ---------------------------------------------------------------------------
__global__ __launch_bounds__(256) void cvt_bf16_kernel(const float* __restrict__ in,
                                                       unsigned short* __restrict__ out,
                                                       int n4) {
    int i = blockIdx.x * blockDim.x + threadIdx.x;
    if (i < n4) {
        float4 v = ((const float4*)in)[i];
        uint2 p;
        p.x = (unsigned int)f2bf(v.x) | ((unsigned int)f2bf(v.y) << 16);
        p.y = (unsigned int)f2bf(v.z) | ((unsigned int)f2bf(v.w) << 16);
        ((uint2*)out)[i] = p;
    }
}

// ---------------------------------------------------------------------------
// Tiled bf16 GEMM: C = A(MxK, row-major bf16) * Bm(KxN, row-major bf16).
// Block tile 128x64, 8 waves, each wave 32x32 (2x2 WMMA 16x16x32 tiles).
// EPI 0: bf16 row-major out.  EPI 1: bf16 TRANSPOSED out (C[n*ldcT + m]).
// EPI 2: f32 out with reference triu mask: col > (row % 1024) -> -1e9.
// blockIdx.z -> (b = z/H, h = z%H) selecting A/B/C batch slices via strides.
// ---------------------------------------------------------------------------
template <int EPI>
__global__ __launch_bounds__(256) void gemm_bf16_kernel(
    const unsigned short* __restrict__ A,
    const unsigned short* __restrict__ Bm,
    void* __restrict__ Cv,
    int Nn, int Kk,
    long long aStride, long long bStride, long long cStride, int ldcT)
{
    __shared__ unsigned short ldsA[128 * 32];   // 8 KB
    __shared__ unsigned short ldsBt[64 * 32];   // 4 KB, transposed [n][k]

    const int z = blockIdx.z;
    const int b = z / H_;
    const int h = z % H_;
    A  += (size_t)b * aStride;
    Bm += (size_t)h * bStride;

    const int tid  = threadIdx.x;
    const int wid  = tid >> 5;
    const int lane = tid & 31;
    const int half = lane >> 4;
    const int r    = lane & 15;

    const int m0 = blockIdx.x * 128;
    const int n0 = blockIdx.y * 64;
    const int wm = (wid >> 1) * 32;
    const int wn = (wid & 1) * 32;

    v8f acc00 = vzero8(), acc01 = vzero8(), acc10 = vzero8(), acc11 = vzero8();

    for (int k0 = 0; k0 < Kk; k0 += 32) {
        // Stage A tile 128x32 (coalesced 16B loads)
        #pragma unroll
        for (int l = tid; l < 128 * 4; l += 256) {
            int ar = l >> 2, ch = (l & 3) * 8;
            uint4 v = *(const uint4*)(A + (size_t)(m0 + ar) * Kk + k0 + ch);
            *(uint4*)&ldsA[ar * 32 + ch] = v;
        }
        // Stage B tile 32x64 transposed into ldsBt[n][k]
        #pragma unroll
        for (int l = tid; l < 32 * 16; l += 256) {
            int k = l >> 4, nc = (l & 15) * 4;
            uint2 v = *(const uint2*)(Bm + (size_t)(k0 + k) * Nn + n0 + nc);
            ldsBt[(nc + 0) * 32 + k] = (unsigned short)(v.x & 0xFFFFu);
            ldsBt[(nc + 1) * 32 + k] = (unsigned short)(v.x >> 16);
            ldsBt[(nc + 2) * 32 + k] = (unsigned short)(v.y & 0xFFFFu);
            ldsBt[(nc + 3) * 32 + k] = (unsigned short)(v.y >> 16);
        }
        // Prefetch next K-tile while this one is consumed (global_prefetch_b8).
        if (k0 + 32 < Kk) {
            __builtin_prefetch(A + (size_t)(m0 + (tid >> 1)) * Kk + (k0 + 32), 0, 0);
            __builtin_prefetch(Bm + (size_t)(k0 + 32 + (tid & 31)) * Nn + n0, 0, 0);
        }
        __syncthreads();

        v16bf a0 = load_a_frag(ldsA, wm + r,      half, 32);
        v16bf a1 = load_a_frag(ldsA, wm + 16 + r, half, 32);
        v16bf b0 = load_b_frag_contig(ldsBt, wn + r,      half, 32);
        v16bf b1 = load_b_frag_contig(ldsBt, wn + 16 + r, half, 32);

        acc00 = wmma_bf16(a0, b0, acc00);
        acc01 = wmma_bf16(a0, b1, acc01);
        acc10 = wmma_bf16(a1, b0, acc10);
        acc11 = wmma_bf16(a1, b1, acc11);
        __syncthreads();
    }

    // Epilogue. C layout: VGPR i -> row (i + 8*half), col = lane&15.
    #pragma unroll
    for (int ai = 0; ai < 2; ++ai) {
        #pragma unroll
        for (int bi = 0; bi < 2; ++bi) {
            v8f acc = (ai == 0) ? ((bi == 0) ? acc00 : acc01)
                                : ((bi == 0) ? acc10 : acc11);
            #pragma unroll
            for (int i = 0; i < 8; ++i) {
                int row = m0 + wm + ai * 16 + i + 8 * half;
                int col = n0 + wn + bi * 16 + r;
                float v = acc[i];
                if (EPI == 0) {
                    unsigned short* C = (unsigned short*)Cv + (size_t)z * cStride;
                    C[(size_t)row * Nn + col] = f2bf(v);
                } else if (EPI == 1) {
                    unsigned short* C = (unsigned short*)Cv + (size_t)z * cStride;
                    C[(size_t)col * ldcT + row] = f2bf(v);
                } else {
                    float* C = (float*)Cv;
                    C[(size_t)row * Nn + col] =
                        (col > (row & (N_ - 1))) ? -1.0e9f : v;
                }
            }
        }
    }
}

// ---------------------------------------------------------------------------
// Flash attention: one wave per 16 query rows, key tiles of 32.
// Q,K bf16 row-major [z][N][D]; Vt bf16 transposed [z][DV][N];
// out Hc bf16 [b][n][h*DV + dv].
// ---------------------------------------------------------------------------
__global__ __launch_bounds__(256) void flash_attn_kernel(
    const unsigned short* __restrict__ Q,
    const unsigned short* __restrict__ Km,
    const unsigned short* __restrict__ Vt,
    unsigned short* __restrict__ Hc)
{
    __shared__ unsigned short ldsP[8 * 16 * 32];  // per-wave P staging, 8 KB

    const int tid  = threadIdx.x;
    const int wid  = tid >> 5;
    const int lane = tid & 31;
    const int half = lane >> 4;
    const int r    = lane & 15;

    const int q0 = blockIdx.x * 128 + wid * 16;
    const int z  = blockIdx.y;
    const int b  = z >> 4;          // z = b*H + h
    const int h  = z & 15;

    const unsigned short* Qp = Q  + (size_t)z * N_ * D_;
    const unsigned short* Kp = Km + (size_t)z * N_ * D_;
    const unsigned short* Vp = Vt + (size_t)z * DV_ * N_;
    unsigned short* myP = &ldsP[wid * 16 * 32];

    const float scale = 0.03125f;   // 1/sqrt(1024)

    float mrow[8], lrow[8];
    #pragma unroll
    for (int i = 0; i < 8; ++i) { mrow[i] = -3.0e38f; lrow[i] = 0.0f; }
    v8f o[4];
    #pragma unroll
    for (int t = 0; t < 4; ++t) o[t] = vzero8();

    for (int j0 = 0; j0 < N_; j0 += 32) {
        // --- S(16x32) = Q_tile(16xD) * K_j(Dx32)^T ------------------------
        v8f s0 = vzero8(), s1 = vzero8();
        for (int k0 = 0; k0 < D_; k0 += 32) {
            // K offset folded into the base pointers (Qp+k0 / Kp+k0).
            v16bf qf  = load_a_frag(Qp + k0, q0 + r, half, D_);
            // Q*K^T: B-columns are rows of K -> contiguous along D.
            v16bf kf0 = load_b_frag_contig(Kp + k0, j0 + r,      half, D_);
            v16bf kf1 = load_b_frag_contig(Kp + k0, j0 + 16 + r, half, D_);
            s0 = wmma_bf16(qf, kf0, s0);
            s1 = wmma_bf16(qf, kf1, s1);
        }

        // --- online softmax (rows live in 16-lane halves) ------------------
        #pragma unroll
        for (int i = 0; i < 8; ++i) {
            float a = s0[i] * scale;
            float c = s1[i] * scale;
            float mx = fmaxf(a, c);
            #pragma unroll
            for (int msk = 8; msk >= 1; msk >>= 1)
                mx = fmaxf(mx, __shfl_xor(mx, msk, 16));
            float mnew  = fmaxf(mrow[i], mx);
            float alpha = __expf(mrow[i] - mnew);
            float p0 = __expf(a - mnew);
            float p1 = __expf(c - mnew);
            float rs = p0 + p1;
            #pragma unroll
            for (int msk = 8; msk >= 1; msk >>= 1)
                rs += __shfl_xor(rs, msk, 16);
            lrow[i] = lrow[i] * alpha + rs;
            mrow[i] = mnew;
            #pragma unroll
            for (int t = 0; t < 4; ++t) o[t][i] *= alpha;
            int prow = i + 8 * half;
            myP[prow * 32 + r]      = f2bf(p0);
            myP[prow * 32 + 16 + r] = f2bf(p1);
        }

        // --- O += P(16x32) * V_j(32x64); Vt gives contiguous B-columns ----
        v16bf pf = load_a_frag(myP, r, half, 32);
        #pragma unroll
        for (int t = 0; t < 4; ++t) {
            v16bf vf = load_b_frag_contig(Vp + j0, t * 16 + r, half, N_);
            o[t] = wmma_bf16(pf, vf, o[t]);
        }
    }

    // --- normalize and store Hc[b][row][h*64 + dv] ------------------------
    #pragma unroll
    for (int t = 0; t < 4; ++t) {
        #pragma unroll
        for (int i = 0; i < 8; ++i) {
            int row = q0 + i + 8 * half;
            int col = h * DV_ + t * 16 + r;
            float v = o[t][i] / lrow[i];
            Hc[((size_t)b * N_ + row) * (H_ * DV_) + col] = f2bf(v);
        }
    }
}

// ---------------------------------------------------------------------------
// Host-side launcher
// ---------------------------------------------------------------------------
extern "C" void kernel_launch(void* const* d_in, const int* in_sizes, int n_in,
                              void* d_out, int out_size, void* d_ws, size_t ws_size,
                              hipStream_t stream) {
    (void)in_sizes; (void)n_in; (void)out_size; (void)ws_size;

    const float* X  = (const float*)d_in[0];
    const float* Wq = (const float*)d_in[1];
    const float* Wk = (const float*)d_in[2];
    const float* Wv = (const float*)d_in[3];
    const float* Wo = (const float*)d_in[4];

    unsigned short* w = (unsigned short*)d_ws;
    size_t off = 0;
    unsigned short* Xb  = w + off; off += (size_t)B_ * N_ * D_;       // 4 M
    unsigned short* Wqb = w + off; off += (size_t)H_ * D_ * D_;       // 16 M
    unsigned short* Wkb = w + off; off += (size_t)H_ * D_ * D_;       // 16 M
    unsigned short* Wvb = w + off; off += (size_t)H_ * D_ * DV_;      // 1 M
    unsigned short* Wob = w + off; off += (size_t)(H_ * DV_) * D_;    // 1 M
    unsigned short* Qb  = w + off; off += (size_t)B_ * H_ * N_ * D_;  // 64 M
    unsigned short* Kb  = w + off; off += (size_t)B_ * H_ * N_ * D_;  // 64 M
    unsigned short* Vtb = w + off; off += (size_t)B_ * H_ * DV_ * N_; // 4 M
    unsigned short* Hcb = w + off; off += (size_t)B_ * N_ * H_ * DV_; // 4 M

    auto cvt = [&](const float* src, unsigned short* dst, size_t n) {
        int n4 = (int)(n / 4);
        cvt_bf16_kernel<<<(n4 + 255) / 256, 256, 0, stream>>>(src, dst, n4);
    };
    cvt(X,  Xb,  (size_t)B_ * N_ * D_);
    cvt(Wq, Wqb, (size_t)H_ * D_ * D_);
    cvt(Wk, Wkb, (size_t)H_ * D_ * D_);
    cvt(Wv, Wvb, (size_t)H_ * D_ * DV_);
    cvt(Wo, Wob, (size_t)(H_ * DV_) * D_);

    const long long sXb = (long long)N_ * D_;        // per-batch A stride
    const long long sW  = (long long)D_ * D_;        // per-head W_q/W_k stride
    const long long sQK = (long long)N_ * D_;        // per-(b,h) Q/K stride

    // Q = X[b] * W_q[h]
    gemm_bf16_kernel<0><<<dim3(N_ / 128, D_ / 64, B_ * H_), 256, 0, stream>>>(
        Xb, Wqb, Qb, D_, D_, sXb, sW, sQK, 0);
    // K = X[b] * W_k[h]
    gemm_bf16_kernel<0><<<dim3(N_ / 128, D_ / 64, B_ * H_), 256, 0, stream>>>(
        Xb, Wkb, Kb, D_, D_, sXb, sW, sQK, 0);
    // Vt = (X[b] * W_v[h])^T  -> [z][DV][N]
    gemm_bf16_kernel<1><<<dim3(N_ / 128, 1, B_ * H_), 256, 0, stream>>>(
        Xb, Wvb, Vtb, DV_, D_, sXb, (long long)D_ * DV_, (long long)DV_ * N_, N_);

    // Flash attention -> Hc [B][N][H*DV]
    flash_attn_kernel<<<dim3(N_ / 128, B_ * H_), 256, 0, stream>>>(Qb, Kb, Vtb, Hcb);

    // Y = Hc * W_o with triu(d > n) -> -1e9 mask in epilogue (f32 out)
    gemm_bf16_kernel<2><<<dim3((B_ * N_) / 128, D_ / 64, 1), 256, 0, stream>>>(
        Hcb, Wob, d_out, D_, D_, 0, 0, 0, 0);
}
